// Coma_67001489817917
// MI455X (gfx1250) — matrix operand order repaired and checked by reading, not compiled, not run on hardware
//
#include <hip/hip_runtime.h>
#include <hip/hip_bf16.h>
#include <math.h>

// ---------------------------------------------------------------------------
// CDNA5 / gfx1250 implementation of the CoMA graph-conv VAE forward pass.
// Dense einsums -> v_wmma_f32_16x16x32_bf16 (f32 accum), full-tile fast path
// with b128 A-fragment loads; branchless masked generic path for ragged K/N.
// Sparse cheb propagation -> coalesced f32 global atomics (L2-resident).
// Activations laid out (N, B, F): WMMA rows = n*16+b (multiple of 16), edge
// gathers/scatters touch contiguous B*F runs.
// ---------------------------------------------------------------------------

typedef __attribute__((ext_vector_type(16))) __bf16 bf16x16;
typedef __attribute__((ext_vector_type(8)))  float  f32x8;

#define WPB 4                    // waves per block in the WMMA GEMM
static constexpr int Bb = 16;
static constexpr int NL[5]   = {20000, 5000, 1250, 313, 79};
static constexpr int EL[5]   = {120000, 30000, 7500, 1878, 474};
static constexpr int FILT[6] = {3, 32, 32, 64, 64, 64};
static constexpr int LAT_IN  = 79 * 64;   // 5056
static constexpr int ZC      = 64 + 2;    // z + cond

static inline unsigned cdiv(long a, long b) { return (unsigned)((a + b - 1) / b); }

// ------------------------------- elementwise -------------------------------

__global__ void k_scale_copy(float* __restrict__ dst, const float* __restrict__ src,
                             float alpha, long n) {
  long i = (long)blockIdx.x * blockDim.x + threadIdx.x;
  if (i < n) dst[i] = src ? alpha * src[i] : 0.0f;
}

__global__ void k_deg_count(const int* __restrict__ row, float* deg, int E) {
  int e = blockIdx.x * blockDim.x + threadIdx.x;
  if (e < E) atomicAdd(&deg[row[e]], 1.0f);
}

__global__ void k_edge_norm(const int* __restrict__ row, const int* __restrict__ col,
                            const float* __restrict__ deg, float* __restrict__ norm,
                            int E) {
  int e = blockIdx.x * blockDim.x + threadIdx.x;
  if (e >= E) return;
  float dr = deg[row[e]], dc = deg[col[e]];
  float a = dr > 0.f ? rsqrtf(dr) : 0.f;
  float b = dc > 0.f ? rsqrtf(dc) : 0.f;
  norm[e] = a * b;
}

// out[tgt[e]] += scale * norm[e] * in[src[e]]   over B*F contiguous channels
__global__ void k_prop(const int* __restrict__ src, const int* __restrict__ tgt,
                       const float* __restrict__ norm, int E,
                       const float* __restrict__ in, float* __restrict__ out,
                       int BF, float scale) {
  long total  = (long)E * BF;
  long stride = (long)gridDim.x * blockDim.x;
  for (long i = (long)blockIdx.x * blockDim.x + threadIdx.x; i < total; i += stride) {
    int e = (int)(i / BF);
    int j = (int)(i - (long)e * BF);
    atomicAdd(&out[(long)tgt[e] * BF + j], scale * norm[e] * in[(long)src[e] * BF + j]);
  }
}

__global__ void k_gather_nodes(const float* __restrict__ in, float* __restrict__ out,
                               const int* __restrict__ idx, int Nn, int BF) {
  long total = (long)Nn * BF;
  long i = (long)blockIdx.x * blockDim.x + threadIdx.x;
  if (i >= total) return;
  int n = (int)(i / BF);
  int j = (int)(i - (long)n * BF);
  out[i] = in[(long)idx[n] * BF + j];
}

__global__ void k_upsample(const float* __restrict__ in, float* __restrict__ out,
                           const int* __restrict__ us_idx, const float* __restrict__ us_w,
                           int Nn, int BF) {
  long total = (long)Nn * BF;
  long i = (long)blockIdx.x * blockDim.x + threadIdx.x;
  if (i >= total) return;
  int n = (int)(i / BF);
  int j = (int)(i - (long)n * BF);
  float acc = 0.0f;
#pragma unroll
  for (int r = 0; r < 3; ++r)
    acc += us_w[n * 3 + r] * in[(long)us_idx[n * 3 + r] * BF + j];
  out[i] = acc;
}

// (B, N, F) flat -> (N, B, F)
__global__ void k_bnf_to_nbf(const float* __restrict__ in, float* __restrict__ out,
                             int Nn, int F) {
  long total = (long)Bb * Nn * F;
  long i = (long)blockIdx.x * blockDim.x + threadIdx.x;
  if (i >= total) return;
  int f = (int)(i % F);
  int n = (int)((i / F) % Nn);
  int b = (int)(i / ((long)F * Nn));
  out[((long)n * Bb + b) * F + f] = in[i];
}

// (N, B, F) -> (B, N, F) flat
__global__ void k_nbf_to_bnf(const float* __restrict__ in, float* __restrict__ out,
                             int Nn, int F) {
  long total = (long)Bb * Nn * F;
  long i = (long)blockIdx.x * blockDim.x + threadIdx.x;
  if (i >= total) return;
  int f = (int)(i % F);
  int n = (int)((i / F) % Nn);
  int b = (int)(i / ((long)F * Nn));
  out[i] = in[((long)n * Bb + b) * F + f];
}

__device__ __forceinline__ unsigned mixhash(unsigned x) {
  x ^= x >> 16; x *= 0x7feb352du;
  x ^= x >> 15; x *= 0x846ca68bu;
  x ^= x >> 16;
  return x;
}

// z = mu + eps*exp(0.5*logvar); concat label -> dec_in (16 x 66)
__global__ void k_make_decin(const float* __restrict__ mu, const float* __restrict__ lv,
                             const float* __restrict__ label, float* __restrict__ dec_in) {
  int i = blockIdx.x * blockDim.x + threadIdx.x;
  if (i >= Bb * ZC) return;
  int b = i / ZC, j = i % ZC;
  if (j < 64) {
    int t = b * 64 + j;
    unsigned h1 = mixhash(0x9e3779b9u ^ (unsigned)t);
    unsigned h2 = mixhash(0x85ebca6bu + (unsigned)t * 0x27d4eb2fu);
    float u1 = ((h1 >> 8) + 1.0f) * (1.0f / 16777217.0f);
    float u2 = (h2 >> 8) * (1.0f / 16777216.0f);
    float eps = sqrtf(-2.0f * logf(u1)) * cosf(6.28318530718f * u2);
    dec_in[i] = mu[t] + eps * expf(0.5f * lv[t]);
  } else {
    dec_in[i] = label[b * 2 + (j - 64)];
  }
}

// ------------------------------ WMMA GEMM ----------------------------------
// Out(M,N) {=, +=} A(M,K) @ W(K,N)  [+ bias][ReLU], bf16 inputs, f32 accum.
// One wave computes one 16x16 D tile via v_wmma_f32_16x16x32_bf16.
// Fragment layouts per CDNA5 ISA 7.12.2 (wave32):
//   A 16x32: M = lane%16; lane half selects K-octet: contiguous runs
//            [k0+8*hi, +7] (VGPRs 0-3) and [k0+16+8*hi, +7] (VGPRs 4-7)
//            -> two 32B runs per lane -> 4x global_load_b128 in FULL path.
//   B 32x16: N = lane%16; VGPR j -> K = 8*(lane/16) + 2j+{0,1} + (j>=4 ? 8 : 0)
//   C/D    : N = lane%16; VGPR r -> M = r + 8*(lane/16)
// FULL = (K%32==0 && N%16==0): no guards. Generic path: clamp-address +
// multiply-by-mask (v_cndmask), never exec-branching around loads.
template <bool FULL>
__global__ __launch_bounds__(WPB * 32)
void k_gemm_wmma_bf16(const float* __restrict__ A, const float* __restrict__ W,
                      float* __restrict__ Out, const float* __restrict__ bias,
                      int M, int K, int N, int acc, int relu) {
  const int lane = threadIdx.x & 31;
  const int wave = threadIdx.x >> 5;
  const int m0   = (blockIdx.x * WPB + wave) * 16;
  const int n0   = blockIdx.y * 16;
  if (m0 >= M) return;
  const int col = lane & 15;
  const int hi  = lane >> 4;
  const int nn  = n0 + col;
  const int   nc    = FULL ? nn : (nn < N ? nn : 0);
  const float nmask = FULL ? 1.0f : (nn < N ? 1.0f : 0.0f);

  f32x8 c;
  if (acc) {
#pragma unroll
    for (int r = 0; r < 8; ++r)
      c[r] = Out[(long)(m0 + r + 8 * hi) * N + nc] * nmask;
  } else {
    float bv = bias ? bias[nc] * nmask : 0.0f;
#pragma unroll
    for (int r = 0; r < 8; ++r) c[r] = bv;
  }

  const float* __restrict__ arow = A + (long)(m0 + col) * K;  // lane's A row
  for (int k0 = 0; k0 < K; k0 += 32) {
    bf16x16 a, b;
    if (FULL) {
      const float4 a0 = *(const float4*)(arow + k0 + 8 * hi);
      const float4 a1 = *(const float4*)(arow + k0 + 8 * hi + 4);
      const float4 a2 = *(const float4*)(arow + k0 + 16 + 8 * hi);
      const float4 a3 = *(const float4*)(arow + k0 + 16 + 8 * hi + 4);
      a[0]  = (__bf16)a0.x; a[1]  = (__bf16)a0.y; a[2]  = (__bf16)a0.z; a[3]  = (__bf16)a0.w;
      a[4]  = (__bf16)a1.x; a[5]  = (__bf16)a1.y; a[6]  = (__bf16)a1.z; a[7]  = (__bf16)a1.w;
      a[8]  = (__bf16)a2.x; a[9]  = (__bf16)a2.y; a[10] = (__bf16)a2.z; a[11] = (__bf16)a2.w;
      a[12] = (__bf16)a3.x; a[13] = (__bf16)a3.y; a[14] = (__bf16)a3.z; a[15] = (__bf16)a3.w;
#pragma unroll
      for (int j = 0; j < 8; ++j) {
        int kw = k0 + ((j < 4) ? (2 * j + 8 * hi) : (16 + 2 * (j - 4) + 8 * hi));
        b[2 * j]     = (__bf16)W[(long)kw * N + nn];
        b[2 * j + 1] = (__bf16)W[(long)(kw + 1) * N + nn];
      }
    } else {
#pragma unroll
      for (int j = 0; j < 8; ++j) {
        int kb = k0 + ((j < 4) ? (2 * j + 8 * hi) : (16 + 2 * (j - 4) + 8 * hi));
        int   kb0 = (kb     < K) ? kb     : 0;
        int   kb1 = (kb + 1 < K) ? kb + 1 : 0;
        float m0f = (kb     < K) ? 1.0f : 0.0f;
        float m1f = (kb + 1 < K) ? 1.0f : 0.0f;
        a[2 * j]     = (__bf16)(arow[kb0] * m0f);
        a[2 * j + 1] = (__bf16)(arow[kb1] * m1f);
        int kw = k0 + ((j < 4) ? (2 * j + 8 * hi) : (16 + 2 * (j - 4) + 8 * hi));
        int   kw0 = (kw     < K) ? kw     : 0;
        int   kw1 = (kw + 1 < K) ? kw + 1 : 0;
        float w0f = (kw     < K) ? nmask : 0.0f;
        float w1f = (kw + 1 < K) ? nmask : 0.0f;
        b[2 * j]     = (__bf16)(W[(long)kw0 * N + nc] * w0f);
        b[2 * j + 1] = (__bf16)(W[(long)kw1 * N + nc] * w1f);
      }
    }
    c = __builtin_amdgcn_wmma_f32_16x16x32_bf16(false, a, false, b,
                                                (short)0, c, false, false);
  }

#pragma unroll
  for (int r = 0; r < 8; ++r) {
    float v = c[r];
    if (relu) v = fmaxf(v, 0.0f);
    if (FULL) {
      Out[(long)(m0 + r + 8 * hi) * N + nn] = v;
    } else if (nn < N) {
      Out[(long)(m0 + r + 8 * hi) * N + nn] = v;
    }
  }
}

// ------------------------------- host side ---------------------------------

static void run_gemm(const float* A, const float* W, float* Out, const float* bias,
                     int M, int K, int N, int acc, int relu, hipStream_t s) {
  dim3 grid(cdiv(M / 16, WPB), cdiv(N, 16));
  if ((K % 32 == 0) && (N % 16 == 0))
    k_gemm_wmma_bf16<true><<<grid, WPB * 32, 0, s>>>(A, W, Out, bias, M, K, N, acc, relu);
  else
    k_gemm_wmma_bf16<false><<<grid, WPB * 32, 0, s>>>(A, W, Out, bias, M, K, N, acc, relu);
}

// One Chebyshev conv layer: O = sum_k T_k(L~) H @ W[k] (+bias)(+relu)
// H is clobbered. S1,S2 scratch. All buffers (Nn*Bb*Fi / *Fo) floats.
static void cheb(const int* edges, const float* normv, int E, int Nn,
                 float* H, float* S1, float* S2, float* O,
                 const float* W, const float* bias, int Fi, int Fo, int relu,
                 hipStream_t s) {
  const int* esrc = edges;
  const int* etgt = edges + E;
  const int M  = Nn * Bb;
  const int BF = Bb * Fi;
  const long nelem = (long)M * Fi;
  const long ptot  = (long)E * BF;
  const int TB = 256;

  run_gemm(H, W, O, bias, M, Fi, Fo, 0, 0, s);                       // T0 term + bias
  k_scale_copy<<<cdiv(nelem, TB), TB, 0, s>>>(S1, nullptr, 0.f, nelem);
  k_prop<<<cdiv(ptot, TB), TB, 0, s>>>(esrc, etgt, normv, E, H, S1, BF, 1.0f);
  run_gemm(S1, W + (long)1 * Fi * Fo, O, nullptr, M, Fi, Fo, 1, 0, s); // T1 term

  float* t0 = H; float* t1 = S1; float* fr = S2;
  for (int k = 2; k < 6; ++k) {
    k_scale_copy<<<cdiv(nelem, TB), TB, 0, s>>>(fr, t0, -1.0f, nelem); // fr = -T_{k-2}
    k_prop<<<cdiv(ptot, TB), TB, 0, s>>>(esrc, etgt, normv, E, t1, fr, BF, 2.0f);
    run_gemm(fr, W + (long)k * Fi * Fo, O, nullptr, M, Fi, Fo, 1, (k == 5) ? relu : 0, s);
    float* tmp = t0; t0 = t1; t1 = fr; fr = tmp;
  }
}

extern "C" void kernel_launch(void* const* d_in, const int* in_sizes, int n_in,
                              void* d_out_v, int out_size, void* d_ws, size_t ws_size,
                              hipStream_t stream) {
  // ---- inputs, in setup_inputs() dict order ----
  const float* x     = (const float*)d_in[0];
  const float* label = (const float*)d_in[1];
  const int* edges[5];
  for (int i = 0; i < 5; ++i) edges[i] = (const int*)d_in[2 + i];
  const int* ds_idx[4]; const int* us_idx[4]; const float* us_w[4];
  for (int i = 0; i < 4; ++i) {
    ds_idx[i] = (const int*)d_in[7 + 3 * i];
    us_idx[i] = (const int*)d_in[8 + 3 * i];
    us_w[i]   = (const float*)d_in[9 + 3 * i];
  }
  const float* enc_w[4]; const float* enc_b[4];
  for (int i = 0; i < 4; ++i) {
    enc_w[i] = (const float*)d_in[19 + 2 * i];
    enc_b[i] = (const float*)d_in[20 + 2 * i];
  }
  const float* dec_w[5];
  for (int i = 0; i < 5; ++i) dec_w[i] = (const float*)d_in[27 + i];
  const float* dec_b[4];
  for (int i = 0; i < 4; ++i) dec_b[i] = (const float*)d_in[32 + i];
  const float* enc_mu_w  = (const float*)d_in[36];
  const float* enc_mu_b  = (const float*)d_in[37];
  const float* enc_lv_w  = (const float*)d_in[38];
  const float* enc_lv_b  = (const float*)d_in[39];
  const float* dec_lin_w = (const float*)d_in[40];
  const float* dec_lin_b = (const float*)d_in[41];

  float* out    = (float*)d_out_v;        // (B*N0, 3) flat, b-major
  float* mu_out = out + (long)Bb * NL[0] * 3;
  float* lv_out = mu_out + Bb * 64;

  // ---- workspace carve ----
  float* ws = (float*)d_ws;
  size_t off = 0;
  float* normv[5];
  for (int i = 0; i < 5; ++i) { normv[i] = ws + off; off += EL[i]; }
  float* deg    = ws + off; off += NL[0];
  float* lat    = ws + off; off += (long)Bb * LAT_IN;
  float* lat2   = ws + off; off += (long)Bb * LAT_IN;
  float* dec_in = ws + off; off += Bb * ZC;
  const long CAP = (long)NL[0] * Bb * 32;   // largest activation: level0 @ F=32
  float* p0 = ws + off; off += CAP;
  float* p1 = ws + off; off += CAP;
  float* p2 = ws + off; off += CAP;
  float* p3 = ws + off; off += CAP;
  (void)ws_size; (void)in_sizes; (void)n_in; (void)out_size;

  const int TB = 256;

  // ---- 1) symmetric edge norms per level ----
  for (int l = 0; l < 5; ++l) {
    k_scale_copy<<<cdiv(NL[l], TB), TB, 0, stream>>>(deg, nullptr, 0.f, NL[l]);
    k_deg_count<<<cdiv(EL[l], TB), TB, 0, stream>>>(edges[l], deg, EL[l]);
    k_edge_norm<<<cdiv(EL[l], TB), TB, 0, stream>>>(edges[l], edges[l] + EL[l],
                                                    deg, normv[l], EL[l]);
  }

  // ---- 2) x: (B,N0,3) -> (N0,B,3) ----
  k_bnf_to_nbf<<<cdiv((long)Bb * NL[0] * 3, TB), TB, 0, stream>>>(x, p0, NL[0], 3);

  // ---- 3) encoder: 4x (cheb + relu, downsample) ----
  for (int i = 0; i < 4; ++i) {
    cheb(edges[i], normv[i], EL[i], NL[i], p0, p1, p2, p3,
         enc_w[i], enc_b[i], FILT[i], FILT[i + 1], 1, stream);
    long tot = (long)NL[i + 1] * Bb * FILT[i + 1];
    k_gather_nodes<<<cdiv(tot, TB), TB, 0, stream>>>(p3, p0, ds_idx[i],
                                                     NL[i + 1], Bb * FILT[i + 1]);
  }

  // ---- 4) latent: mu / logvar / z / dec_lin ----
  k_nbf_to_bnf<<<cdiv((long)Bb * 79 * 64, TB), TB, 0, stream>>>(p0, lat, 79, 64);
  run_gemm(lat, enc_mu_w, mu_out, enc_mu_b, Bb, LAT_IN, 64, 0, 0, stream);
  run_gemm(lat, enc_lv_w, lv_out, enc_lv_b, Bb, LAT_IN, 64, 0, 0, stream);
  k_make_decin<<<cdiv(Bb * ZC, TB), TB, 0, stream>>>(mu_out, lv_out, label, dec_in);
  run_gemm(dec_in, dec_lin_w, lat2, dec_lin_b, Bb, ZC, LAT_IN, 0, 1, stream);
  k_bnf_to_nbf<<<cdiv((long)Bb * 79 * 64, TB), TB, 0, stream>>>(lat2, p0, 79, 64);

  // ---- 5) decoder: 4x (upsample, cheb + relu) ----
  for (int i = 0; i < 4; ++i) {
    int lvl = 3 - i;
    int Fi = FILT[5 - i], Fo = FILT[4 - i];
    long tot = (long)NL[lvl] * Bb * Fi;
    k_upsample<<<cdiv(tot, TB), TB, 0, stream>>>(p0, p1, us_idx[lvl], us_w[lvl],
                                                 NL[lvl], Bb * Fi);
    cheb(edges[lvl], normv[lvl], EL[lvl], NL[lvl], p1, p2, p3, p0,
         dec_w[i], dec_b[i], Fi, Fo, 1, stream);
  }

  // ---- 6) final cheb (32 -> 3, no bias/relu) and write-out ----
  cheb(edges[0], normv[0], EL[0], NL[0], p0, p1, p2, p3,
       dec_w[4], nullptr, 32, 3, 0, stream);
  k_nbf_to_bnf<<<cdiv((long)Bb * NL[0] * 3, TB), TB, 0, stream>>>(p3, out, NL[0], 3);
}
